// MambaBlock_34205119545698
// MI455X (gfx1250) — compile-verified, matrix operand order checked by reading, not run on hardware
//
#include <hip/hip_runtime.h>
#include <math.h>

#define D_MODEL 1024
#define D_STATE 16
#define D_CONV  4
#define DT_RANK 64
#define D_INNER 2048
#define B_SZ    2
#define L_SEQ   2048
#define M_ROWS  (B_SZ * L_SEQ)          // 4096 rows (b*L + l)
#define N_DBC   (DT_RANK + 2 * D_STATE) // 96

typedef __attribute__((ext_vector_type(16))) __bf16 v16bf;
typedef __attribute__((ext_vector_type(8)))  __bf16 v8bf;
typedef __attribute__((ext_vector_type(8)))  float  v8f;

// ---------------------------------------------------------------------------
// bf16 fragment load (A 16x32 or B 32x16, ISA 7.12.2): per lane the 16
// elements are K[khalf+0..7] and K[khalf+16..23] -> two contiguous 16-byte
// runs -> two global_load_b128, no conversion.
// ---------------------------------------------------------------------------
__device__ __forceinline__ v16bf load_frag_bf(const __bf16* __restrict__ p)
{
    const v8bf lo = *(const v8bf*)(p);
    const v8bf hi = *(const v8bf*)(p + 16);
    v16bf f;
#pragma unroll
    for (int i = 0; i < 8; ++i) { f[i] = lo[i]; f[i + 8] = hi[i]; }
    return f;
}

// ---------------------------------------------------------------------------
// Generic C[M,N] = A[M,K] * W[N,K]^T, A/W pre-staged in bf16, f32 accumulate.
// Block = 128 threads = 4 waves; wave w owns rows [by*64 + w*16, +16),
// cols [bx*16*WN, +16*WN).
// MODE 1: fused softplus(v + bias[col]).  BFOUT: also store bf16 mirror.
// All loads of a k-step are issued before the WMMAs (MLP / s_clause).
// ---------------------------------------------------------------------------
template <int WN, int MODE, bool BFOUT>
__global__ __launch_bounds__(128) void gemm_wmma_bf16(
    const __bf16* __restrict__ A, const __bf16* __restrict__ W,
    float* __restrict__ C, const float* __restrict__ bias,
    __bf16* __restrict__ Cbf,
    int K, int lda, int ldw, int ldc)
{
    const int lane    = threadIdx.x & 31;
    const int half16  = lane & 15;
    const int khalf   = (lane & 16) ? 8 : 0;
    const int wave    = threadIdx.x >> 5;
    const int rowbase = blockIdx.y * 64 + wave * 16;
    const int colbase = blockIdx.x * (16 * WN);

    const __bf16* pa = A + (size_t)(rowbase + half16) * lda + khalf;
    const __bf16* pw = W + (size_t)(colbase + half16) * ldw + khalf;
    const size_t wstep = (size_t)16 * ldw;   // stride between the WN col-tiles

    v8f acc[WN];
#pragma unroll
    for (int j = 0; j < WN; ++j) acc[j] = {};

    for (int k = 0; k < K; k += 32) {
        // ---- issue ALL fragment loads for this k-step first ----
        const v16bf a = load_frag_bf(pa + k);
        v16bf b[WN];
#pragma unroll
        for (int j = 0; j < WN; ++j) b[j] = load_frag_bf(pw + (size_t)j * wstep + k);

        // ---- WMMAs ----
#pragma unroll
        for (int j = 0; j < WN; ++j)
            acc[j] = __builtin_amdgcn_wmma_f32_16x16x32_bf16(
                false, a, false, b[j], (short)0, acc[j], false, false);
    }

    const int rrow = rowbase + ((lane & 16) ? 8 : 0);
#pragma unroll
    for (int j = 0; j < WN; ++j) {
        const int ncol = colbase + j * 16 + half16;
#pragma unroll
        for (int r = 0; r < 8; ++r) {
            float v = acc[j][r];
            if (MODE == 1) { // softplus(v + bias)
                const float t = v + bias[ncol];
                v = (t > 20.0f) ? t : log1pf(__expf(t));
            }
            C[(size_t)(rrow + r) * ldc + ncol] = v;
            if (BFOUT)
                Cbf[(size_t)(rrow + r) * ldc + ncol] = (__bf16)v;
        }
    }
}

// ---------------------------------------------------------------------------
// One-time fp32 -> bf16 conversion, 8 elements per thread (b128 in, b128 out)
// ---------------------------------------------------------------------------
__global__ __launch_bounds__(256) void cvt_f32_to_bf16(
    const float* __restrict__ in, __bf16* __restrict__ outp, int n8)
{
    const int i = blockIdx.x * 256 + threadIdx.x;
    if (i >= n8) return;
    const float4 a = ((const float4*)in)[2 * i];
    const float4 b = ((const float4*)in)[2 * i + 1];
    v8bf o;
    o[0] = (__bf16)a.x; o[1] = (__bf16)a.y; o[2] = (__bf16)a.z; o[3] = (__bf16)a.w;
    o[4] = (__bf16)b.x; o[5] = (__bf16)b.y; o[6] = (__bf16)b.z; o[7] = (__bf16)b.w;
    ((v8bf*)outp)[i] = o;
}

// ---------------------------------------------------------------------------
// Depthwise causal conv (k=4) + bias + SiLU; output staged as bf16 for the
// x_proj GEMM (scan converts on load).  xc is cols [0,ED) of xz.
// ---------------------------------------------------------------------------
__global__ __launch_bounds__(256) void conv_silu_kernel(
    const float* __restrict__ xz, const float* __restrict__ cw,
    const float* __restrict__ cb, __bf16* __restrict__ xabf)
{
    const int idx = blockIdx.x * 256 + threadIdx.x;     // over B*L*ED
    const int e  = idx & (D_INNER - 1);
    const int bl = idx >> 11;                           // b*L + l
    const int l  = bl & (L_SEQ - 1);
    float acc = cb[e];
#pragma unroll
    for (int i = 0; i < D_CONV; ++i) {
        const int li = l - (D_CONV - 1) + i;
        if (li >= 0)
            acc += cw[e * D_CONV + i] *
                   xz[(size_t)(bl - (D_CONV - 1) + i) * (2 * D_INNER) + e];
    }
    xabf[(size_t)bl * D_INNER + e] = (__bf16)(acc / (1.0f + __expf(-acc))); // silu
}

// ---------------------------------------------------------------------------
// Selective scan + D-skip + SiLU(z) gate. One thread per (b, e) channel,
// h[16] in VGPRs; B_t/C_t broadcast through LDS each step.
// Writes yg directly as bf16 for the out_proj GEMM.
// ---------------------------------------------------------------------------
__global__ __launch_bounds__(256) void scan_gate_kernel(
    const __bf16* __restrict__ xabf, const float* __restrict__ delta,
    const float* __restrict__ dbc, const float* __restrict__ xz,
    const float* __restrict__ A_log, const float* __restrict__ Dp,
    __bf16* __restrict__ ygbf)
{
    const int b = blockIdx.y;
    const int e = blockIdx.x * 256 + threadIdx.x;
    __shared__ float sB[D_STATE];
    __shared__ float sC[D_STATE];

    float a[D_STATE], h[D_STATE];
#pragma unroll
    for (int n = 0; n < D_STATE; ++n) {
        a[n] = -__expf(A_log[e * D_STATE + n]);
        h[n] = 0.0f;
    }
    const float dskip = Dp[e];

    for (int l = 0; l < L_SEQ; ++l) {
        const size_t rl = (size_t)(b * L_SEQ + l);
        __syncthreads();
        if (threadIdx.x < 2 * D_STATE) {
            const float v = dbc[rl * N_DBC + DT_RANK + threadIdx.x];
            if (threadIdx.x < D_STATE) sB[threadIdx.x] = v;
            else                       sC[threadIdx.x - D_STATE] = v;
        }
        __syncthreads();

        const float d  = delta[rl * D_INNER + e];
        const float xv = (float)xabf[rl * D_INNER + e];
        const float dx = d * xv;
        float y = 0.0f;
#pragma unroll
        for (int n = 0; n < D_STATE; ++n) {
            h[n] = __expf(d * a[n]) * h[n] + dx * sB[n];
            y += h[n] * sC[n];
        }
        y += xv * dskip;
        const float zv = xz[rl * (2 * D_INNER) + D_INNER + e];
        y *= zv / (1.0f + __expf(-zv));                 // * silu(z)
        ygbf[rl * D_INNER + e] = (__bf16)y;
    }
}

// ---------------------------------------------------------------------------
extern "C" void kernel_launch(void* const* d_in, const int* in_sizes, int n_in,
                              void* d_out, int out_size, void* d_ws, size_t ws_size,
                              hipStream_t stream)
{
    (void)in_sizes; (void)n_in; (void)out_size; (void)ws_size;
    const float* x     = (const float*)d_in[0]; // (B,L,1024)
    const float* w_in  = (const float*)d_in[1]; // (4096,1024)
    const float* cw    = (const float*)d_in[2]; // (2048,1,4)
    const float* cb    = (const float*)d_in[3]; // (2048)
    const float* w_xp  = (const float*)d_in[4]; // (96,2048)
    const float* w_dt  = (const float*)d_in[5]; // (2048,64)
    const float* b_dt  = (const float*)d_in[6]; // (2048)
    const float* alog  = (const float*)d_in[7]; // (2048,16)
    const float* dpar  = (const float*)d_in[8]; // (2048)
    const float* w_out = (const float*)d_in[9]; // (1024,2048)
    float* out = (float*)d_out;                 // (B,L,1024) f32

    // ---- workspace carve-up (fp32 first, then 16B-aligned bf16 region) ----
    const size_t N_XZ  = (size_t)M_ROWS * 2 * D_INNER;  // 16,777,216
    const size_t N_DB  = (size_t)M_ROWS * N_DBC;        //    393,216
    const size_t N_DEL = (size_t)M_ROWS * D_INNER;      //  8,388,608
    const size_t N_X   = (size_t)M_ROWS * D_MODEL;      //  4,194,304
    const size_t N_WIN = (size_t)(2 * D_INNER) * D_MODEL;
    const size_t N_WXP = (size_t)N_DBC * D_INNER;
    const size_t N_WDT = (size_t)D_INNER * DT_RANK;
    const size_t N_WO  = (size_t)D_MODEL * D_INNER;
    const size_t N_XA  = (size_t)M_ROWS * D_INNER;

    float* xz    = (float*)d_ws;
    float* dbc   = xz  + N_XZ;
    float* delta = dbc + N_DB;
    __bf16* xbf    = (__bf16*)(delta + N_DEL);
    __bf16* winbf  = xbf    + N_X;
    __bf16* wxpbf  = winbf  + N_WIN;
    __bf16* wdtbf  = wxpbf  + N_WXP;
    __bf16* woutbf = wdtbf  + N_WDT;
    __bf16* xabf   = woutbf + N_WO;
    __bf16* dbcbf  = xabf   + N_XA;
    __bf16* ygbf   = dbcbf  + N_DB;

    const dim3 blk(128);

    // 0) one-time bf16 staging of x and all GEMM weights
    cvt_f32_to_bf16<<<(N_X   / 8 + 255) / 256, 256, 0, stream>>>(x,     xbf,    (int)(N_X / 8));
    cvt_f32_to_bf16<<<(N_WIN / 8 + 255) / 256, 256, 0, stream>>>(w_in,  winbf,  (int)(N_WIN / 8));
    cvt_f32_to_bf16<<<(N_WXP / 8 + 255) / 256, 256, 0, stream>>>(w_xp,  wxpbf,  (int)(N_WXP / 8));
    cvt_f32_to_bf16<<<(N_WDT / 8 + 255) / 256, 256, 0, stream>>>(w_dt,  wdtbf,  (int)(N_WDT / 8));
    cvt_f32_to_bf16<<<(N_WO  / 8 + 255) / 256, 256, 0, stream>>>(w_out, woutbf, (int)(N_WO / 8));

    // 1) in_proj: xz = x @ w_in^T   (M=4096, N=4096, K=1024)
    gemm_wmma_bf16<4, 0, false><<<dim3((2 * D_INNER) / 64, M_ROWS / 64), blk, 0, stream>>>(
        xbf, winbf, xz, nullptr, nullptr, D_MODEL, D_MODEL, D_MODEL, 2 * D_INNER);

    // 2) depthwise causal conv + SiLU -> xa (bf16)
    conv_silu_kernel<<<(B_SZ * L_SEQ * D_INNER) / 256, 256, 0, stream>>>(xz, cw, cb, xabf);

    // 3) x_proj: dbc = xa @ w_xp^T  (M=4096, N=96, K=2048); fp32 + bf16 mirror
    gemm_wmma_bf16<1, 0, true><<<dim3(N_DBC / 16, M_ROWS / 64), blk, 0, stream>>>(
        xabf, wxpbf, dbc, nullptr, dbcbf, D_INNER, D_INNER, D_INNER, N_DBC);

    // 4) dt_proj + softplus: delta = softplus(dlt @ w_dt^T + b_dt)
    //    dlt = dbcbf[:, 0:64] (lda = 96), M=4096, N=2048, K=64
    gemm_wmma_bf16<4, 1, false><<<dim3(D_INNER / 64, M_ROWS / 64), blk, 0, stream>>>(
        dbcbf, wdtbf, delta, b_dt, nullptr, DT_RANK, N_DBC, DT_RANK, D_INNER);

    // 5) selective scan + skip + gate -> yg (bf16)
    scan_gate_kernel<<<dim3(D_INNER / 256, B_SZ), 256, 0, stream>>>(
        xabf, delta, dbc, xz, alog, dpar, ygbf);

    // 6) out_proj: out = yg @ w_out^T (M=4096, N=1024, K=2048)
    gemm_wmma_bf16<4, 0, false><<<dim3(D_MODEL / 64, M_ROWS / 64), blk, 0, stream>>>(
        ygbf, woutbf, out, nullptr, nullptr, D_INNER, D_INNER, D_INNER, D_MODEL);
}
